// EmbeddingSearchLayer_39178691674884
// MI455X (gfx1250) — matrix operand order repaired and checked by reading, not compiled, not run on hardware
//
#include <hip/hip_runtime.h>
#include <hip/hip_bf16.h>

// ---------------------------------------------------------------------------
// EmbeddingSearch (argmin_n ||q_m - v_n||) for M=2048, N=131072, D=256 (fp32)
//   dist^2 = ||v||^2 - 2 q.v   (+||q||^2 constant per row -> dropped)
// Cross term via split-bf16 WMMA: q.v ~= qh.vh + qh.vl + ql.vh (fp32 accum)
//
// Two paths chosen at launch by ws_size:
//  FAST: one-time kernel pre-splits V and Q into packed bf16 hi/lo arrays in
//        ws (~141 MB). GEMM kernel stages copy-only, keeps the full-K Q tile
//        resident in LDS, and double-buffers V chunks (105 KB LDS total,
//        one barrier per K-chunk).
//  FALLBACK: convert-on-the-fly in the GEMM kernel (37 KB LDS), used when ws
//        is too small for the split copies.
// ---------------------------------------------------------------------------

typedef __attribute__((ext_vector_type(16))) __bf16 v16bf;
typedef __attribute__((ext_vector_type(8)))  __bf16 v8bf;
typedef __attribute__((ext_vector_type(8)))  float  v8f;

#define M_Q   2048
#define N_V   131072
#define D_K   256
#define BM    64
#define BN    64
#define KC    64                      // K-chunk staged in LDS (V side)
#define NITER 8                       // V-tiles processed per block
#define TCH   (NITER * 4)             // 32 K-chunks per block pass
#define NG    (N_V / (BN * NITER))    // 256 = grid.x
#define MT    (M_Q / BM)              // 32  = grid.y
#define LSTR  72                      // KC + 8 pad (bf16): 144B row stride
#define QSTR  264                     // D_K + 8 pad (bf16): 528B row stride

static __device__ __forceinline__ unsigned short bf16_rtne(float x) {
  unsigned u = __float_as_uint(x);
  return (unsigned short)((u + 0x7FFFu + ((u >> 16) & 1u)) >> 16);
}

union FragB16 {
  v16bf v;
  struct { v8bf lo, hi; } p;
};

// ---------------------------------------------------------------------------
// ||v||^2 per vector (fp32, exact). One wave32 per vector.
// ---------------------------------------------------------------------------
__global__ __launch_bounds__(256) void vsq_kernel(const float* __restrict__ V,
                                                  float* __restrict__ vsq) {
  const int wave = threadIdx.x >> 5, lane = threadIdx.x & 31;
  const int v = blockIdx.x * 8 + wave;
  const float4* p = (const float4*)(V + (size_t)v * D_K + lane * 8);
  float4 a = p[0], b = p[1];
  float ss = a.x * a.x + a.y * a.y + a.z * a.z + a.w * a.w
           + b.x * b.x + b.y * b.y + b.z * b.z + b.w * b.w;
  #pragma unroll
  for (int off = 16; off; off >>= 1) ss += __shfl_xor(ss, off, 32);
  if (lane == 0) vsq[v] = ss;
}

// ---------------------------------------------------------------------------
// One-time fp32 -> packed bf16 (hi=trunc, lo=rtne residual). Pure bandwidth.
// ---------------------------------------------------------------------------
__global__ __launch_bounds__(256) void conv_split_kernel(
    const float4* __restrict__ src, uint2* __restrict__ dh,
    uint2* __restrict__ dl, int n4) {
  int i = blockIdx.x * 256 + threadIdx.x;
  if (i >= n4) return;
  float4 f = src[i];
  unsigned u0 = __float_as_uint(f.x), u1 = __float_as_uint(f.y);
  unsigned u2 = __float_as_uint(f.z), u3 = __float_as_uint(f.w);
  uint2 h, l;
  h.x = (u0 >> 16) | (u1 & 0xFFFF0000u);
  h.y = (u2 >> 16) | (u3 & 0xFFFF0000u);
  float l0 = f.x - __uint_as_float(u0 & 0xFFFF0000u);
  float l1 = f.y - __uint_as_float(u1 & 0xFFFF0000u);
  float l2 = f.z - __uint_as_float(u2 & 0xFFFF0000u);
  float l3 = f.w - __uint_as_float(u3 & 0xFFFF0000u);
  l.x = (unsigned)bf16_rtne(l0) | ((unsigned)bf16_rtne(l1) << 16);
  l.y = (unsigned)bf16_rtne(l2) | ((unsigned)bf16_rtne(l3) << 16);
  dh[i] = h;
  dl[i] = l;
}

// ---------------------------------------------------------------------------
// Shared epilogue: in-register running argmin -> per-block partials.
// C layout: N = lane%16, M = vgpr slot + 8*(lane>=16).
// ---------------------------------------------------------------------------
static __device__ __forceinline__ void argmin_epilogue(
    float best[8], int bidx[8], float rv[2][BM], int ri[2][BM],
    int tid, int lr, int lh, int wm, int wn, int m0, int bx,
    float* __restrict__ pval, int* __restrict__ pidx) {
  #pragma unroll
  for (int r = 0; r < 8; ++r) {
    #pragma unroll
    for (int off = 1; off < 16; off <<= 1) {
      float ov = __shfl_xor(best[r], off, 32);
      int   oi = __shfl_xor(bidx[r], off, 32);
      if (ov < best[r] || (ov == best[r] && oi < bidx[r])) {
        best[r] = ov; bidx[r] = oi;
      }
    }
  }
  if (lr == 0) {
    const int mrow = wm * 16 + lh * 8;
    #pragma unroll
    for (int r = 0; r < 8; ++r) { rv[wn][mrow + r] = best[r]; ri[wn][mrow + r] = bidx[r]; }
  }
  __syncthreads();
  if (tid < BM) {
    float v0 = rv[0][tid]; int i0 = ri[0][tid];
    float v1 = rv[1][tid]; int i1 = ri[1][tid];
    if (v1 < v0 || (v1 == v0 && i1 < i0)) { v0 = v1; i0 = i1; }
    pval[(size_t)(m0 + tid) * NG + bx] = v0;
    pidx[(size_t)(m0 + tid) * NG + bx] = i0;
  }
}

// ---------------------------------------------------------------------------
// FAST PATH: pre-split bf16 inputs; full-K Q resident in LDS; copy-only,
// double-buffered V staging per K-chunk (one barrier per chunk).
// 8 waves = 4(M) x 2(N); wave tile = 16M x 32N.
// LDS: Qh+Ql 67.6KB + 2x(Vh+Vl) 36.9KB + 1KB reduce = ~105KB.
// ---------------------------------------------------------------------------
__global__ __launch_bounds__(256) void nn_pre_kernel(
    const unsigned int* __restrict__ Qh, const unsigned int* __restrict__ Ql,
    const unsigned int* __restrict__ Vh, const unsigned int* __restrict__ Vl,
    const float* __restrict__ vsq, float* __restrict__ pval,
    int* __restrict__ pidx) {
  __shared__ unsigned short sQh[BM * QSTR], sQl[BM * QSTR];
  __shared__ unsigned short sVh[2][BN * LSTR], sVl[2][BN * LSTR];
  __shared__ float rv[2][BM];
  __shared__ int   ri[2][BM];

  const int tid  = threadIdx.x;
  const int lane = tid & 31, wave = tid >> 5;
  const int wm = wave >> 1, wn = wave & 1;
  const int lr = lane & 15, lh = lane >> 4;

  const int m0     = blockIdx.y * BM;
  const int ngroup = blockIdx.x * (BN * NITER);

  // Stage chunk t of V (tile t>>2, K-chunk t&3) into ping-pong buffer `buf`.
  auto stageV = [&](int t, int buf) {
    const int nb = ngroup + (t >> 2) * BN;
    const int kc = t & 3;
    unsigned int* lVh = (unsigned int*)sVh[buf];
    unsigned int* lVl = (unsigned int*)sVl[buf];
    #pragma unroll
    for (int i = 0; i < 2; ++i) {
      const int idx = tid + i * 256;      // uint4 units over [64][8]
      const int row = idx >> 3;
      const int c   = (idx & 7) << 2;     // uint column within chunk
      const size_t g = (size_t)(nb + row) * (D_K / 2) + kc * (KC / 2) + c;
      uint4 ah = *(const uint4*)&Vh[g];
      uint4 al = *(const uint4*)&Vl[g];
      *(uint4*)&lVh[row * (LSTR / 2) + c] = ah;
      *(uint4*)&lVl[row * (LSTR / 2) + c] = al;
      if (t + 4 < TCH)                    // warm GL2 one tile ahead
        __builtin_prefetch(&Vh[g + (size_t)BN * (D_K / 2)], 0, 0);
    }
  };

  // Stage full-K Q tile once (packed-uint copies, no conversion).
  {
    unsigned int* lQh = (unsigned int*)sQh;
    unsigned int* lQl = (unsigned int*)sQl;
    #pragma unroll
    for (int i = 0; i < 8; ++i) {
      const int idx = tid + i * 256;        // uint4 units over [64][32]
      const int row = idx >> 5;
      const int c   = (idx & 31) << 2;      // uint column
      uint4 ah = *(const uint4*)&Qh[(size_t)(m0 + row) * (D_K / 2) + c];
      uint4 al = *(const uint4*)&Ql[(size_t)(m0 + row) * (D_K / 2) + c];
      *(uint4*)&lQh[row * (QSTR / 2) + c] = ah;
      *(uint4*)&lQl[row * (QSTR / 2) + c] = al;
    }
  }

  float best[8];
  int   bidx[8];
  #pragma unroll
  for (int r = 0; r < 8; ++r) { best[r] = 3.4e38f; bidx[r] = 0; }

  v8f c0 = {};
  v8f c1 = {};

  stageV(0, 0);
  __syncthreads();

  for (int t = 0; t < TCH; ++t) {
    if (t + 1 < TCH) stageV(t + 1, (t + 1) & 1);   // overlap with compute

    const int buf = t & 1;
    const int kc  = t & 3;
    const unsigned short* bVh = sVh[buf];
    const unsigned short* bVl = sVl[buf];

    #pragma unroll
    for (int ks = 0; ks < KC; ks += 32) {
      const int arow = (wm * 16 + lr) * QSTR + kc * KC + ks + lh * 8;
      FragB16 ah, al;
      ah.p.lo = *(const v8bf*)&sQh[arow];
      ah.p.hi = *(const v8bf*)&sQh[arow + 16];
      al.p.lo = *(const v8bf*)&sQl[arow];
      al.p.hi = *(const v8bf*)&sQl[arow + 16];
      #pragma unroll
      for (int nsub = 0; nsub < 2; ++nsub) {
        const int brow = (wn * 32 + nsub * 16 + lr) * LSTR + ks + lh * 8;
        FragB16 bh, bl;
        bh.p.lo = *(const v8bf*)&bVh[brow];
        bh.p.hi = *(const v8bf*)&bVh[brow + 16];
        bl.p.lo = *(const v8bf*)&bVl[brow];
        bl.p.hi = *(const v8bf*)&bVl[brow + 16];
        v8f& c = nsub ? c1 : c0;
        c = __builtin_amdgcn_wmma_f32_16x16x32_bf16(false, ah.v, false, bh.v,
                                                    (short)0, c, false, false);
        c = __builtin_amdgcn_wmma_f32_16x16x32_bf16(false, ah.v, false, bl.v,
                                                    (short)0, c, false, false);
        c = __builtin_amdgcn_wmma_f32_16x16x32_bf16(false, al.v, false, bh.v,
                                                    (short)0, c, false, false);
      }
    }

    if ((t & 3) == 3) {                  // finished a V tile: fused scoring
      const int nb  = ngroup + (t >> 2) * BN;
      const int gn0 = nb + wn * 32 + lr;
      const int gn1 = gn0 + 16;
      const float vs0 = vsq[gn0];
      const float vs1 = vsq[gn1];
      #pragma unroll
      for (int r = 0; r < 8; ++r) {
        float s0 = vs0 - 2.0f * c0[r];
        float s1 = vs1 - 2.0f * c1[r];
        if (s0 < best[r]) { best[r] = s0; bidx[r] = gn0; }
        if (s1 < best[r]) { best[r] = s1; bidx[r] = gn1; }
      }
      c0 = (v8f){};
      c1 = (v8f){};
    }

    __syncthreads();                     // staging(t+1) done; readers done
  }

  argmin_epilogue(best, bidx, rv, ri, tid, lr, lh, wm, wn, m0, blockIdx.x,
                  pval, pidx);
}

// ---------------------------------------------------------------------------
// FALLBACK: convert-on-the-fly (37 KB LDS), used when ws can't hold splits.
// ---------------------------------------------------------------------------
__global__ __launch_bounds__(256) void nn_kernel(const float* __restrict__ Q,
                                                 const float* __restrict__ V,
                                                 const float* __restrict__ vsq,
                                                 float* __restrict__ pval,
                                                 int*   __restrict__ pidx) {
  __shared__ unsigned short sQh[BM * LSTR], sQl[BM * LSTR];
  __shared__ unsigned short sVh[BN * LSTR], sVl[BN * LSTR];
  __shared__ float rv[2][BM];
  __shared__ int   ri[2][BM];

  const int tid  = threadIdx.x;
  const int lane = tid & 31, wave = tid >> 5;
  const int wm = wave >> 1, wn = wave & 1;
  const int lr = lane & 15, lh = lane >> 4;

  const int m0     = blockIdx.y * BM;
  const int ngroup = blockIdx.x * (BN * NITER);

  float best[8];
  int   bidx[8];
  #pragma unroll
  for (int r = 0; r < 8; ++r) { best[r] = 3.4e38f; bidx[r] = 0; }

  for (int it = 0; it < NITER; ++it) {
    const int nb = ngroup + it * BN;
    v8f c0 = {};
    v8f c1 = {};

    for (int kc = 0; kc < D_K; kc += KC) {
      __syncthreads();
      #pragma unroll
      for (int i = 0; i < 4; ++i) {
        const int idx = tid + i * 256;
        const int row = idx >> 4;
        const int c4  = (idx & 15) << 2;
        float4 fq = *(const float4*)(Q + (size_t)(m0 + row) * D_K + kc + c4);
        float4 fv = *(const float4*)(V + (size_t)(nb + row) * D_K + kc + c4);
        const int base = row * LSTR + c4;
        float qe[4] = {fq.x, fq.y, fq.z, fq.w};
        float ve[4] = {fv.x, fv.y, fv.z, fv.w};
        #pragma unroll
        for (int j = 0; j < 4; ++j) {
          unsigned uq = __float_as_uint(qe[j]);
          unsigned uv = __float_as_uint(ve[j]);
          sQh[base + j] = (unsigned short)(uq >> 16);   // trunc hi
          sVh[base + j] = (unsigned short)(uv >> 16);
          sQl[base + j] = bf16_rtne(qe[j] - __uint_as_float(uq & 0xFFFF0000u));
          sVl[base + j] = bf16_rtne(ve[j] - __uint_as_float(uv & 0xFFFF0000u));
        }
        if (it + 1 < NITER)
          __builtin_prefetch(V + (size_t)(nb + BN + row) * D_K + kc + c4, 0, 0);
      }
      __syncthreads();

      #pragma unroll
      for (int ks = 0; ks < KC; ks += 32) {
        const int arow = (wm * 16 + lr) * LSTR + ks + lh * 8;
        FragB16 ah, al;
        ah.p.lo = *(const v8bf*)&sQh[arow];
        ah.p.hi = *(const v8bf*)&sQh[arow + 16];
        al.p.lo = *(const v8bf*)&sQl[arow];
        al.p.hi = *(const v8bf*)&sQl[arow + 16];
        #pragma unroll
        for (int nsub = 0; nsub < 2; ++nsub) {
          const int brow = (wn * 32 + nsub * 16 + lr) * LSTR + ks + lh * 8;
          FragB16 bh, bl;
          bh.p.lo = *(const v8bf*)&sVh[brow];
          bh.p.hi = *(const v8bf*)&sVh[brow + 16];
          bl.p.lo = *(const v8bf*)&sVl[brow];
          bl.p.hi = *(const v8bf*)&sVl[brow + 16];
          v8f& c = nsub ? c1 : c0;
          c = __builtin_amdgcn_wmma_f32_16x16x32_bf16(false, ah.v, false, bh.v,
                                                      (short)0, c, false, false);
          c = __builtin_amdgcn_wmma_f32_16x16x32_bf16(false, ah.v, false, bl.v,
                                                      (short)0, c, false, false);
          c = __builtin_amdgcn_wmma_f32_16x16x32_bf16(false, al.v, false, bh.v,
                                                      (short)0, c, false, false);
        }
      }
    }

    const int gn0 = nb + wn * 32 + lr;
    const int gn1 = gn0 + 16;
    const float vs0 = vsq[gn0];
    const float vs1 = vsq[gn1];
    #pragma unroll
    for (int r = 0; r < 8; ++r) {
      float s0 = vs0 - 2.0f * c0[r];
      float s1 = vs1 - 2.0f * c1[r];
      if (s0 < best[r]) { best[r] = s0; bidx[r] = gn0; }
      if (s1 < best[r]) { best[r] = s1; bidx[r] = gn1; }
    }
  }

  argmin_epilogue(best, bidx, rv, ri, tid, lr, lh, wm, wn, m0, blockIdx.x,
                  pval, pidx);
}

// ---------------------------------------------------------------------------
// Reduce NG=256 partials per query -> int32 argmin index.
// ---------------------------------------------------------------------------
__global__ __launch_bounds__(NG) void argmin_reduce_kernel(
    const float* __restrict__ pval, const int* __restrict__ pidx,
    int* __restrict__ out) {
  __shared__ float sv[NG];
  __shared__ int   si[NG];
  const int m = blockIdx.x, t = threadIdx.x;
  sv[t] = pval[(size_t)m * NG + t];
  si[t] = pidx[(size_t)m * NG + t];
  __syncthreads();
  for (int s = NG >> 1; s > 0; s >>= 1) {
    if (t < s) {
      float a = sv[t], b = sv[t + s];
      int ia = si[t], ib = si[t + s];
      if (b < a || (b == a && ib < ia)) { sv[t] = b; si[t] = ib; }
    }
    __syncthreads();
  }
  if (t == 0) out[m] = si[0];
}

// ---------------------------------------------------------------------------
extern "C" void kernel_launch(void* const* d_in, const int* in_sizes, int n_in,
                              void* d_out, int out_size, void* d_ws, size_t ws_size,
                              hipStream_t stream) {
  const float* Q = (const float*)d_in[0];   // [M, D] fp32
  const float* V = (const float*)d_in[1];   // [N, D] fp32

  // Workspace layout: vsq[N] | pval[M*NG] | pidx[M*NG] | Vh | Vl | Qh | Ql
  float* vsq  = (float*)d_ws;
  float* pval = vsq + N_V;
  int*   pidx = (int*)(pval + (size_t)M_Q * NG);
  unsigned int* Vh = (unsigned int*)(pidx + (size_t)M_Q * NG);
  unsigned int* Vl = Vh + (size_t)N_V * (D_K / 2);
  unsigned int* Qh = Vl + (size_t)N_V * (D_K / 2);
  unsigned int* Ql = Qh + (size_t)M_Q * (D_K / 2);
  int* out = (int*)d_out;

  const size_t need = (size_t)(Ql + (size_t)M_Q * (D_K / 2)) - (size_t)d_ws;

  vsq_kernel<<<N_V / 8, 256, 0, stream>>>(V, vsq);

  if (ws_size >= need) {
    const int nv4 = N_V * (D_K / 4);
    const int nq4 = M_Q * (D_K / 4);
    conv_split_kernel<<<(nv4 + 255) / 256, 256, 0, stream>>>(
        (const float4*)V, (uint2*)Vh, (uint2*)Vl, nv4);
    conv_split_kernel<<<(nq4 + 255) / 256, 256, 0, stream>>>(
        (const float4*)Q, (uint2*)Qh, (uint2*)Ql, nq4);
    nn_pre_kernel<<<dim3(NG, MT), 256, 0, stream>>>(Qh, Ql, Vh, Vl, vsq,
                                                    pval, pidx);
  } else {
    nn_kernel<<<dim3(NG, MT), 256, 0, stream>>>(Q, V, vsq, pval, pidx);
  }

  argmin_reduce_kernel<<<M_Q, NG, 0, stream>>>(pval, pidx, out);
}
